// ScaledDotProductAttention_79310866088116
// MI455X (gfx1250) — matrix operand order repaired and checked by reading, not compiled, not run on hardware
//
#include <hip/hip_runtime.h>
#include <hip/hip_bf16.h>
#include <stdint.h>

// ---------------------------------------------------------------------------
// Masked MHA forward for MI455X (gfx1250, wave32).
// Matmuls: v_wmma_f32_16x16x32_bf16 (f32 accumulate).
// GEMM tile staging: GLOBAL_LOAD_ASYNC_TO_LDS_B128 (ASYNCcnt), double-buffered.
// Attention K/V staging: TENSOR_LOAD_TO_LDS (TDM, TENSORcnt), double-buffered.
// ---------------------------------------------------------------------------

typedef __attribute__((ext_vector_type(16))) __bf16 v16bf;
typedef __attribute__((ext_vector_type(8)))  float  v8f;
typedef __attribute__((ext_vector_type(4)))  uint32_t u32x4;
typedef __attribute__((ext_vector_type(8)))  uint32_t u32x8;

#define B_  2
#define S_  2048
#define D_  1024
#define H_  16
#define HD_ 64
#define M_  (B_ * S_)   // 4096

union FragBF { v16bf v; uint32_t u[8]; };

__device__ __forceinline__ uint16_t f2bf(float f) {
  uint32_t u = __builtin_bit_cast(uint32_t, f);
  u += 0x7FFFu + ((u >> 16) & 1u);          // round-to-nearest-even
  return (uint16_t)(u >> 16);
}

// A-matrix (16x32 bf16) per-lane K index: slot i, lane group g (contiguous: k = 16g + 2i).
__device__ __forceinline__ int kA_idx(int i, int g) {
  return 2 * i + 16 * g;
}

// Low 32 bits of a flat shared pointer == wave-relative LDS byte offset.
__device__ __forceinline__ uint32_t to_lds(const void* p) {
  return (uint32_t)(uintptr_t)p;
}

// 16B per-lane async copy global -> LDS (ASYNCcnt).
__device__ __forceinline__ void async_copy16(uint32_t lds_addr, const void* gaddr) {
  asm volatile("global_load_async_to_lds_b128 %0, %1, off"
               :: "v"(lds_addr), "v"(gaddr) : "memory");
}
__device__ __forceinline__ void wait_async0() {
  asm volatile("s_wait_asynccnt 0x0" ::: "memory");
}

// ---------------------------------------------------------------------------
// TDM: load a 2D tile (tile_w dwords x tile_h rows, row stride stride_dw
// dwords) from global into LDS with one TENSOR_LOAD_TO_LDS.
// D# per ISA 8.3/8.4: group0 {count=1, lds_addr, global_addr[56:0], type=2},
// group1 {data_size=4B, tensor dims == tile dims, dim0 stride}.  2D tensor ->
// descriptor groups 2/3 omitted.  Issue under a uniform wave guard.
// ---------------------------------------------------------------------------
__device__ __forceinline__ void tdm_load_2d(uint32_t lds_addr, const void* gaddr,
                                            uint32_t tile_w_dw, uint32_t tile_h,
                                            uint32_t stride_dw) {
  uint64_t ga = (uint64_t)(uintptr_t)gaddr;
  u32x4 g0;
  g0[0] = 1u;                                           // count=1, user mode
  g0[1] = lds_addr;                                     // LDS byte address
  g0[2] = (uint32_t)ga;                                 // global_addr[31:0]
  g0[3] = (uint32_t)((ga >> 32) & 0x01FFFFFFu) | (2u << 30);  // [56:32],type=2
  u32x8 g1;
  g1[0] = (2u << 16);                                   // data_size=4B
  g1[1] = (tile_w_dw & 0xFFFFu) << 16;                  // tensor_dim0[15:0]
  g1[2] = ((tile_w_dw >> 16) & 0xFFFFu) |               // tensor_dim0[31:16]
          ((tile_h & 0xFFFFu) << 16);                   // tensor_dim1[15:0]
  g1[3] = ((tile_h >> 16) & 0xFFFFu) |                  // tensor_dim1[31:16]
          ((tile_w_dw & 0xFFFFu) << 16);                // tile_dim0
  g1[4] = (tile_h & 0xFFFFu);                           // tile_dim1, tile_dim2=0
  g1[5] = stride_dw;                                    // tensor_dim0_stride lo
  g1[6] = 0u;                                           // stride hi, dim1_stride lo
  g1[7] = 0u;
  asm volatile("tensor_load_to_lds %0, %1" :: "s"(g0), "s"(g1) : "memory");
}

// ---------------------------------------------------------------------------
// Stage 0a: elementwise f32 -> bf16 (activations)
// ---------------------------------------------------------------------------
__global__ void cvt_bf16_kernel(const float* __restrict__ src,
                                uint16_t* __restrict__ dst, int n) {
  int i = blockIdx.x * blockDim.x + threadIdx.x;
  int stride = gridDim.x * blockDim.x;
  for (; i < n; i += stride) dst[i] = f2bf(src[i]);
}

// Stage 0b: f32 [D_][D_] -> bf16 transposed [n][k] (weights)
__global__ void cvt_tr_kernel(const float* __restrict__ src,
                              uint16_t* __restrict__ dst) {
  int idx = blockIdx.x * blockDim.x + threadIdx.x;   // over D_*D_
  int r = idx >> 10, c = idx & (D_ - 1);
  dst[(size_t)c * D_ + r] = f2bf(src[idx]);
}

// ---------------------------------------------------------------------------
// Unified GEMM: C[M_,1024] = A(bf16)[M_,1024] x W^T(bf16 stored [n][k]).
// Block 128 thr (4 waves), tile 128x64, K-step 32, double-buffered async LDS,
// 8 WMMA / wave / K-step.
// OUT: 0 = masked bf16 head layout [B][H][S][HD]   (Q,K)
//      1 = masked bf16 d-major layout [B][H][HD][S] (V)
//      2 = plain f32 [M_][1024]                     (out proj)
// ---------------------------------------------------------------------------
template <int OUT>
__global__ __launch_bounds__(128)
void gemm_kernel(const uint16_t* __restrict__ A, const uint16_t* __restrict__ Wt,
                 const float* __restrict__ ind, void* __restrict__ dstv) {
  __shared__ uint16_t As[2][128 * 32];   // [row][kk]
  __shared__ uint16_t Bs[2][64 * 32];    // [n][kk]

  const int tid  = threadIdx.x;
  const int wave = tid >> 5, lane = tid & 31;
  const int g = lane >> 4, ln = lane & 15;
  const int row0 = blockIdx.x * 128;
  const int col0 = blockIdx.y * 64;

  auto issue = [&](int kk0, int p) {
    #pragma unroll
    for (int e = 0; e < 4; ++e) {                       // A tile: 8KB
      int idx = tid + e * 128;
      int r = idx >> 2, ch = (idx & 3) * 8;
      async_copy16(to_lds(&As[p][r * 32 + ch]),
                   A + (size_t)(row0 + r) * D_ + kk0 + ch);
    }
    #pragma unroll
    for (int e = 0; e < 2; ++e) {                       // B tile: 4KB
      int idx = tid + e * 128;
      int n = idx >> 2, ch = (idx & 3) * 8;
      async_copy16(to_lds(&Bs[p][n * 32 + ch]),
                   Wt + (size_t)(col0 + n) * D_ + kk0 + ch);
    }
  };

  v8f acc[2][4] = {};
  issue(0, 0);

  for (int it = 0; it < D_ / 32; ++it) {
    const int p = it & 1;
    wait_async0();
    __syncthreads();
    if (it + 1 < D_ / 32) issue((it + 1) * 32, p ^ 1);

    FragBF a[2];
    #pragma unroll
    for (int s = 0; s < 2; ++s)
      #pragma unroll
      for (int i = 0; i < 8; ++i)
        a[s].u[i] = *(const uint32_t*)
            &As[p][(s * 64 + wave * 16 + ln) * 32 + kA_idx(i, g)];

    #pragma unroll
    for (int t = 0; t < 4; ++t) {
      FragBF b;
      #pragma unroll
      for (int i = 0; i < 8; ++i)
        b.u[i] = *(const uint32_t*)&Bs[p][(t * 16 + ln) * 32 + 2 * i + 16 * g];
      #pragma unroll
      for (int s = 0; s < 2; ++s)
        acc[s][t] = __builtin_amdgcn_wmma_f32_16x16x32_bf16(
            false, a[s].v, false, b.v, (short)0, acc[s][t], false, false);
    }
  }

  #pragma unroll
  for (int s = 0; s < 2; ++s)
    #pragma unroll
    for (int t = 0; t < 4; ++t)
      #pragma unroll
      for (int i = 0; i < 8; ++i) {
        int row = row0 + s * 64 + wave * 16 + i + 8 * g;   // D: row = i + 8g
        int col = col0 + t * 16 + ln;
        if (OUT == 2) {
          ((float*)dstv)[(size_t)row * D_ + col] = acc[s][t][i];
        } else {
          float v = acc[s][t][i] * ind[row];
          int b = row >> 11, sr = row & (S_ - 1);
          int h = col >> 6,  d  = col & 63;
          size_t o = (OUT == 0)
              ? (((size_t)(b * H_ + h)) * S_ + sr) * HD_ + d     // [BH][S][HD]
              : (((size_t)(b * H_ + h)) * HD_ + d) * S_ + sr;    // [BH][HD][S]
          ((uint16_t*)dstv)[o] = f2bf(v);
        }
      }
}

// ---------------------------------------------------------------------------
// Flash attention per (b,h,64 q rows).  4 waves x 16 q rows.
// K/V tiles staged by wave 0 via TDM (one TENSOR_LOAD_TO_LDS per tile),
// double-buffered; s_wait_tensorcnt + workgroup barrier publishes the tile.
// Per 32-key step: 4 WMMA QK^T, online softmax (ind_k folded in), 4 WMMA PV.
// K stored [BH][S][HD]; V stored d-major [BH][HD][S].
// ---------------------------------------------------------------------------
__global__ __launch_bounds__(128)
void attn_kernel(const uint16_t* __restrict__ Qb, const uint16_t* __restrict__ Kb,
                 const uint16_t* __restrict__ Vb, const float* __restrict__ indk,
                 uint16_t* __restrict__ CTX) {
  __shared__ uint16_t Kt[2][32 * 64];    // [key][d]
  __shared__ uint16_t Vt[2][64 * 32];    // [d][key]
  __shared__ uint16_t Pb[4][16 * 32];    // per-wave P tile

  const int tid = threadIdx.x, wave = tid >> 5, lane = tid & 31;
  const int g = lane >> 4, ln = lane & 15;
  const int bh = blockIdx.y;
  const int b = bh >> 4, h = bh & (H_ - 1);
  const int q0 = blockIdx.x * 64;
  const uint16_t* Qh = Qb + (size_t)bh * S_ * HD_;
  const uint16_t* Kh = Kb + (size_t)bh * S_ * HD_;
  const uint16_t* Vh = Vb + (size_t)bh * HD_ * S_;
  const float*    ih = indk + b * S_;

  // TDM issue (wave 0 only; TDM ignores EXEC, guard is wave-uniform).
  auto issue = [&](int kb0, int p) {
    // K: 32 rows x 128B, contiguous  -> 32dw x 32 rows, stride 32dw
    tdm_load_2d(to_lds(&Kt[p][0]), Kh + (size_t)kb0 * HD_, 32u, 32u, 32u);
    // V: 64 rows x 64B, row stride S_*2 bytes -> 16dw x 64 rows, stride 1024dw
    tdm_load_2d(to_lds(&Vt[p][0]), Vh + kb0, 16u, 64u, (uint32_t)(S_ / 2));
  };

  FragBF aq[2];                                      // Q rows, d 0-31 / 32-63
  {
    int qrow = q0 + wave * 16 + ln;
    const uint32_t* qr = (const uint32_t*)(Qh + (size_t)qrow * HD_);
    #pragma unroll
    for (int f = 0; f < 2; ++f)
      #pragma unroll
      for (int i = 0; i < 8; ++i)
        aq[f].u[i] = qr[(kA_idx(i, g) + 32 * f) >> 1];
  }

  float mrun[8], lrun[8];
  #pragma unroll
  for (int i = 0; i < 8; ++i) { mrun[i] = -3e38f; lrun[i] = 0.f; }
  v8f acc[4] = {};

  if (wave == 0) issue(0, 0);
  for (int it = 0; it < S_ / 32; ++it) {
    const int p = it & 1;
    const int kb0 = it * 32;
    if (wave == 0) __builtin_amdgcn_s_wait_tensorcnt(0);
    __syncthreads();                                  // tile published
    if (wave == 0 && it + 1 < S_ / 32) issue(kb0 + 32, p ^ 1);

    // ---- scores: two 16-key tiles, contraction HD=64 = 2 x K32 WMMA ----
    v8f sc[2];
    #pragma unroll
    for (int t2 = 0; t2 < 2; ++t2) {
      v8f s = {};
      #pragma unroll
      for (int dh = 0; dh < 2; ++dh) {
        FragBF bk;
        #pragma unroll
        for (int i = 0; i < 8; ++i) {
          int dd = 2 * i + 16 * g + 32 * dh;
          bk.u[i] = *(const uint32_t*)&Kt[p][(t2 * 16 + ln) * 64 + dd];
        }
        s = __builtin_amdgcn_wmma_f32_16x16x32_bf16(
            false, aq[dh].v, false, bk.v, (short)0, s, false, false);
      }
      sc[t2] = s;
    }

    // ---- online softmax with key mask ----
    float ind0 = ih[kb0 + ln], ind1 = ih[kb0 + 16 + ln];
    float p0[8], p1[8];
    #pragma unroll
    for (int i = 0; i < 8; ++i) {
      float s0 = sc[0][i] * 0.125f;              // 1/sqrt(HD)
      float s1 = sc[1][i] * 0.125f;
      if (ind0 == 0.f) s0 = -3e38f;
      if (ind1 == 0.f) s1 = -3e38f;
      float rm = fmaxf(s0, s1);
      rm = fmaxf(rm, __shfl_xor(rm, 1, 32));
      rm = fmaxf(rm, __shfl_xor(rm, 2, 32));
      rm = fmaxf(rm, __shfl_xor(rm, 4, 32));
      rm = fmaxf(rm, __shfl_xor(rm, 8, 32));
      float mnew = fmaxf(mrun[i], rm);
      float alpha = __expf(mrun[i] - mnew);
      mrun[i] = mnew;
      float e0 = __expf(s0 - mnew) * ind0;       // exact zero when masked
      float e1 = __expf(s1 - mnew) * ind1;
      float rs = e0 + e1;
      rs += __shfl_xor(rs, 1, 32);
      rs += __shfl_xor(rs, 2, 32);
      rs += __shfl_xor(rs, 4, 32);
      rs += __shfl_xor(rs, 8, 32);
      lrun[i] = lrun[i] * alpha + rs;
      p0[i] = e0; p1[i] = e1;
      #pragma unroll
      for (int t = 0; t < 4; ++t) acc[t][i] = acc[t][i] * alpha;
    }

    // ---- P (D layout) -> A fragment via per-wave LDS tile ----
    uint16_t* pb = Pb[wave];
    #pragma unroll
    for (int i = 0; i < 8; ++i) {
      int r = i + 8 * g;
      pb[r * 32 + ln]      = f2bf(p0[i]);
      pb[r * 32 + 16 + ln] = f2bf(p1[i]);
    }
    asm volatile("s_wait_dscnt 0" ::: "memory");   // intra-wave LDS RAW
    FragBF ap;
    #pragma unroll
    for (int i = 0; i < 8; ++i)
      ap.u[i] = *(const uint32_t*)&pb[ln * 32 + kA_idx(i, g)];

    // ---- PV: 4 WMMA across 64 head dims ----
    #pragma unroll
    for (int t = 0; t < 4; ++t) {
      FragBF bv;
      #pragma unroll
      for (int i = 0; i < 8; ++i)
        bv.u[i] = *(const uint32_t*)&Vt[p][(t * 16 + ln) * 32 + 2 * i + 16 * g];
      acc[t] = __builtin_amdgcn_wmma_f32_16x16x32_bf16(
          false, ap.v, false, bv.v, (short)0, acc[t], false, false);
    }
  }

  // ---- normalize (zero-row guard) and store ctx bf16 [B][S][H*HD] ----
  #pragma unroll
  for (int i = 0; i < 8; ++i) {
    float l = lrun[i];
    float inv = (l > 0.f) ? 1.0f / l : 0.0f;
    int s = q0 + wave * 16 + i + 8 * g;
    #pragma unroll
    for (int t = 0; t < 4; ++t) {
      int col = h * HD_ + t * 16 + ln;
      CTX[((size_t)(b * S_ + s)) * (H_ * HD_) + col] = f2bf(acc[t][i] * inv);
    }
  }
}

// ---------------------------------------------------------------------------
// Workspace (bytes):
//   0   Wq_t | 2MB Wk_t | 4MB Wv_t | 6MB Wo_t          (bf16, transposed)
//   8MB Xq | 16MB Xk | 24MB Xv                          (bf16 activations)
//   32MB Q [BH][S][HD] | 40MB K [BH][S][HD] | 48MB V [BH][HD][S]
//   CTX reuses the Xq slot at 8MB (Xq dead after Q projection). Total 56MB.
// ---------------------------------------------------------------------------
extern "C" void kernel_launch(void* const* d_in, const int* in_sizes, int n_in,
                              void* d_out, int out_size, void* d_ws, size_t ws_size,
                              hipStream_t stream) {
  (void)in_sizes; (void)n_in; (void)out_size; (void)ws_size;

  const float* queries = (const float*)d_in[0];
  const float* keys    = (const float*)d_in[1];
  const float* values  = (const float*)d_in[2];
  const float* ind_q   = (const float*)d_in[3];
  const float* ind_k   = (const float*)d_in[4];
  const float* ind_v   = (const float*)d_in[5];
  const float* Wq      = (const float*)d_in[6];
  const float* Wk      = (const float*)d_in[7];
  const float* Wv      = (const float*)d_in[8];
  const float* Wo      = (const float*)d_in[9];

  uint8_t* ws = (uint8_t*)d_ws;
  const size_t MB = 1u << 20;
  uint16_t* Wq_t = (uint16_t*)(ws + 0 * MB);
  uint16_t* Wk_t = (uint16_t*)(ws + 2 * MB);
  uint16_t* Wv_t = (uint16_t*)(ws + 4 * MB);
  uint16_t* Wo_t = (uint16_t*)(ws + 6 * MB);
  uint16_t* Xq   = (uint16_t*)(ws + 8 * MB);
  uint16_t* Xk   = (uint16_t*)(ws + 16 * MB);
  uint16_t* Xv   = (uint16_t*)(ws + 24 * MB);
  uint16_t* Qb   = (uint16_t*)(ws + 32 * MB);
  uint16_t* Kb   = (uint16_t*)(ws + 40 * MB);
  uint16_t* Vb   = (uint16_t*)(ws + 48 * MB);
  uint16_t* CTX  = (uint16_t*)(ws + 8 * MB);   // overlays Xq (dead by then)

  cvt_tr_kernel<<<dim3(D_ * D_ / 256), 256, 0, stream>>>(Wq, Wq_t);
  cvt_tr_kernel<<<dim3(D_ * D_ / 256), 256, 0, stream>>>(Wk, Wk_t);
  cvt_tr_kernel<<<dim3(D_ * D_ / 256), 256, 0, stream>>>(Wv, Wv_t);
  cvt_tr_kernel<<<dim3(D_ * D_ / 256), 256, 0, stream>>>(Wo, Wo_t);

  cvt_bf16_kernel<<<2048, 256, 0, stream>>>(queries, Xq, M_ * D_);
  cvt_bf16_kernel<<<2048, 256, 0, stream>>>(keys,    Xk, M_ * D_);
  cvt_bf16_kernel<<<2048, 256, 0, stream>>>(values,  Xv, M_ * D_);

  dim3 gp(M_ / 128, D_ / 64);   // 32 x 16 blocks
  gemm_kernel<0><<<gp, 128, 0, stream>>>(Xq, Wq_t, ind_q, Qb);
  gemm_kernel<0><<<gp, 128, 0, stream>>>(Xk, Wk_t, ind_k, Kb);
  gemm_kernel<1><<<gp, 128, 0, stream>>>(Xv, Wv_t, ind_v, Vb);

  dim3 ga(S_ / 64, B_ * H_);    // 32 x 32 blocks
  attn_kernel<<<ga, 128, 0, stream>>>(Qb, Kb, Vb, ind_k, CTX);

  gemm_kernel<2><<<gp, 128, 0, stream>>>(CTX, Wo_t, nullptr, (float*)d_out);
}
